// GatedGraphConvEncoder_2353642078830
// MI455X (gfx1250) — matrix operand order repaired and checked by reading, not compiled, not run on hardware
//
#include <hip/hip_runtime.h>
#include <math.h>

// ---------------- problem constants (match reference) ----------------
#define N_NODES 20000
#define T_TOK   16
#define D_DIM   256
#define B_GR    64
#define E_EDGE  320000
#define N_LAY   3
#define N_GRU   4
#define D3_DIM  768   // 3*D

typedef __attribute__((ext_vector_type(16))) _Float16 v16h;
typedef __attribute__((ext_vector_type(8)))  _Float16 v8h;
typedef __attribute__((ext_vector_type(8)))  float    v8f;

// ---------------- small utility kernels ----------------
__global__ void ggc_fill_f32(float* p, float v, int n) {
    int i = blockIdx.x * blockDim.x + threadIdx.x;
    if (i < n) p[i] = v;
}
__global__ void ggc_fill_i32(int* p, int v, int n) {
    int i = blockIdx.x * blockDim.x + threadIdx.x;
    if (i < n) p[i] = v;
}
__global__ void ggc_fill_u32(unsigned* p, unsigned v, int n) {
    int i = blockIdx.x * blockDim.x + threadIdx.x;
    if (i < n) p[i] = v;
}
// f32 -> f16 convert
__global__ void ggc_cvt16(const float* __restrict__ in, _Float16* __restrict__ out, int n) {
    int i = blockIdx.x * blockDim.x + threadIdx.x;
    if (i < n) out[i] = (_Float16)in[i];
}
// transpose+convert nm [D,D] matrices: WT[m][j*D+k] = W[m][k*D+j]  (f32 -> f16)
__global__ void ggc_prep_wt(const float* __restrict__ W, _Float16* __restrict__ WT, int nm) {
    int idx = blockIdx.x * blockDim.x + threadIdx.x;
    if (idx >= nm * D_DIM * D_DIM) return;
    int m = idx / (D_DIM * D_DIM);
    int r = idx - m * (D_DIM * D_DIM);
    int j = r >> 8, k = r & 255;
    WT[(size_t)m * D_DIM * D_DIM + j * D_DIM + k] =
        (_Float16)W[(size_t)m * D_DIM * D_DIM + k * D_DIM + j];
}

// ---------------- 1) embedding mean-pool:  h[n,d] (+ f16 shadow) ----------------
__global__ void ggc_embed_mean(const int* __restrict__ tok,
                               const float* __restrict__ emb,
                               float* __restrict__ h, _Float16* __restrict__ h16) {
    int idx = blockIdx.x * blockDim.x + threadIdx.x;
    if (idx >= N_NODES * D_DIM) return;
    int n = idx >> 8;            // D = 256
    int d = idx & 255;
    float s = 0.f; int cnt = 0;
    #pragma unroll
    for (int t = 0; t < T_TOK; ++t) {
        int tk = tok[n * T_TOK + t];
        if (tk != 0) { s += emb[(size_t)tk * D_DIM + d]; ++cnt; }
    }
    float v = s / (float)(cnt > 0 ? cnt : 1);
    h[idx] = v;
    h16[idx] = (_Float16)v;
}

// ---------------- 2) WMMA GEMM:  Y[nrows,Dout] = A16[nrows,K] @ B16^T (+bias) ----
// A16 : [nrows, K] f16 row-major.  B16 : [Dout, K] f16 row-major (i.e. B(k,j)=B16[j*K+k]).
// One wave computes a 32x64 tile: 2 M-subtiles x 4 N-subtiles, 8 f32 accumulators.
// Per k-step: 4 b128 A-loads + 8 b128 B-loads feed 8 WMMAs (A pair reused 4x, B 2x).
// nrows % 32 == 0, Dout % 64 == 0, K % 32 == 0.
__global__ __launch_bounds__(256) void ggc_gemm_wmma(
        const _Float16* __restrict__ A16, const _Float16* __restrict__ B16,
        const float* __restrict__ bias, float* __restrict__ Y,
        int nrows, int K, int Dout)
{
    int gwave   = (blockIdx.x * blockDim.x + threadIdx.x) >> 5;
    int lane    = threadIdx.x & 31;
    int ntiles4 = Dout >> 6;                    // 64-wide super-tiles
    int mtiles  = nrows >> 5;                   // 32-row super-tiles
    if (gwave >= mtiles * ntiles4) return;      // uniform per wave -> EXEC all-ones
    int mt = gwave / ntiles4, nt = gwave % ntiles4;
    int lm = lane & 15, lh = lane >> 4;
    int ncol0 = nt << 6;
    int mrow0 = mt << 5;

    const _Float16* arow0 = A16 + (size_t)(mrow0 + lm) * K;        // M-subtile 0
    const _Float16* arow1 = A16 + (size_t)(mrow0 + 16 + lm) * K;   // M-subtile 1
    v8f acc[2][4];
    #pragma unroll
    for (int m = 0; m < 2; ++m)
        #pragma unroll
        for (int t = 0; t < 4; ++t) acc[m][t] = (v8f){};

    for (int kk = 0; kk < K; kk += 32) {
        // A fragments (16x32 f16): halfs 0..7 -> K=kk+8*lh+0..7 ; 8..15 -> +16
        const _Float16* ap0 = arow0 + kk + 8 * lh;
        const _Float16* ap1 = arow1 + kk + 8 * lh;
        v16h a0 = __builtin_shufflevector(*(const v8h*)(ap0), *(const v8h*)(ap0 + 16),
                  0,1,2,3,4,5,6,7,8,9,10,11,12,13,14,15);
        v16h a1 = __builtin_shufflevector(*(const v8h*)(ap1), *(const v8h*)(ap1 + 16),
                  0,1,2,3,4,5,6,7,8,9,10,11,12,13,14,15);
        // B fragments (32x16 f16): halfs 0..7 -> K=kk+8*lh+0..7 ; 8..15 -> +16
        #pragma unroll
        for (int t = 0; t < 4; ++t) {
            const _Float16* bp = B16 + (size_t)(ncol0 + t * 16 + lm) * K + kk + 8 * lh;
            v16h b = __builtin_shufflevector(*(const v8h*)(bp), *(const v8h*)(bp + 16),
                     0,1,2,3,4,5,6,7,8,9,10,11,12,13,14,15);
            acc[0][t] = __builtin_amdgcn_wmma_f32_16x16x32_f16(false, a0, false, b,
                        (short)0, acc[0][t], false, false);
            acc[1][t] = __builtin_amdgcn_wmma_f32_16x16x32_f16(false, a1, false, b,
                        (short)0, acc[1][t], false, false);
        }
    }
    // C/D layout: VGPR r -> row = 8*lh + r ; lane%16 -> col
    #pragma unroll
    for (int t = 0; t < 4; ++t) {
        int col = ncol0 + t * 16 + lm;
        float bv = bias ? bias[col] : 0.f;
        #pragma unroll
        for (int m = 0; m < 2; ++m) {
            #pragma unroll
            for (int r = 0; r < 8; ++r) {
                int mrow = mrow0 + m * 16 + (lh << 3) + r;
                Y[(size_t)mrow * Dout + col] = acc[m][t][r] + bv;
            }
        }
    }
}

// ---------------- 3) edge scatter-add: agg[dst] += hw[src] * emask ---------------
__global__ void ggc_scatter_add(const float* __restrict__ hw,
                                const int* __restrict__ src,
                                const int* __restrict__ dst,
                                const float* __restrict__ emask,
                                float* __restrict__ agg) {
    int idx = blockIdx.x * blockDim.x + threadIdx.x;   // E * (D/4) threads
    if (idx >= E_EDGE * (D_DIM / 4)) return;
    int e = idx >> 6;                 // D/4 = 64 quads per edge
    int q = (idx & 63) << 2;
    float m = emask[e];
    if (m == 0.f) return;
    int s = src[e], t = dst[e];
    const float4 v = *(const float4*)(hw + (size_t)s * D_DIM + q);
    float* ap = agg + (size_t)t * D_DIM + q;
    atomicAdd(ap + 0, v.x * m);
    atomicAdd(ap + 1, v.y * m);
    atomicAdd(ap + 2, v.z * m);
    atomicAdd(ap + 3, v.w * m);
}

// ---------------- 4) GRU elementwise (+ optional ReLU, + f16 shadow) ----------------
__global__ void ggc_gru_elem(const float* __restrict__ gi,
                             const float* __restrict__ gh,
                             float* __restrict__ h, _Float16* __restrict__ h16,
                             int do_relu) {
    int idx = blockIdx.x * blockDim.x + threadIdx.x;
    if (idx >= N_NODES * D_DIM) return;
    int n = idx >> 8, d = idx & 255;
    const float* gin = gi + (size_t)n * D3_DIM;
    const float* ghn = gh + (size_t)n * D3_DIM;
    float ir = gin[d], iz = gin[256 + d], inn = gin[512 + d];
    float hr = ghn[d], hz = ghn[256 + d], hnn = ghn[512 + d];
    float r  = 1.f / (1.f + __expf(-(ir + hr)));
    float z  = 1.f / (1.f + __expf(-(iz + hz)));
    float nn = tanhf(inn + r * hnn);
    float hv = (1.f - z) * nn + z * h[idx];
    if (do_relu) hv = fmaxf(hv, 0.f);
    h[idx] = hv;
    h16[idx] = (_Float16)hv;
}

// ---------------- 5) TopK pooling ----------------
__global__ void ggc_pnorm(const float* __restrict__ p, float* __restrict__ out) {
    __shared__ float sh[256];
    float v = p[threadIdx.x];
    sh[threadIdx.x] = v * v;
    __syncthreads();
    for (int s = 128; s > 0; s >>= 1) {
        if ((int)threadIdx.x < s) sh[threadIdx.x] += sh[threadIdx.x + s];
        __syncthreads();
    }
    if (threadIdx.x == 0) out[0] = sqrtf(sh[0]);
}

// one wave per node: score = tanh((x.p)/||p||); skey = keep ? score : -1e9
__global__ void ggc_score(const float* __restrict__ h, const float* __restrict__ p,
                          const float* __restrict__ pnorm, const int* __restrict__ keep,
                          float* __restrict__ score, float* __restrict__ skey) {
    int gid = blockIdx.x * blockDim.x + threadIdx.x;
    int n = gid >> 5, lane = gid & 31;
    if (n >= N_NODES) return;
    const float* row = h + (size_t)n * D_DIM;
    float s = 0.f;
    for (int d = lane; d < D_DIM; d += 32) s += row[d] * p[d];
    for (int off = 16; off > 0; off >>= 1) s += __shfl_down(s, off, 32);
    if (lane == 0) {
        float sc = tanhf(s / pnorm[0]);
        score[n] = sc;
        skey[n]  = keep[n] ? sc : -1e9f;
    }
}

__global__ void ggc_count(const int* __restrict__ batch, const int* __restrict__ keep,
                          int* __restrict__ counts, int* __restrict__ nvalid) {
    int n = blockIdx.x * blockDim.x + threadIdx.x;
    if (n >= N_NODES) return;
    int b = batch[n];
    atomicAdd(&counts[b], 1);
    if (keep[n]) atomicAdd(&nvalid[b], 1);
}

__global__ void ggc_scan(const int* __restrict__ counts, int* __restrict__ starts) {
    if (blockIdx.x == 0 && threadIdx.x == 0) {
        int acc = 0;
        for (int b = 0; b < B_GR; ++b) { starts[b] = acc; acc += counts[b]; }
    }
}

// rank within graph by (skey desc, index asc), keep top ceil(0.8 * nvalid)
__global__ void ggc_rank(const int* __restrict__ batch, const float* __restrict__ skey,
                         const int* __restrict__ counts, const int* __restrict__ starts,
                         const int* __restrict__ nvalid, const int* __restrict__ keep_in,
                         int* __restrict__ keep_out) {
    int n = blockIdx.x * blockDim.x + threadIdx.x;
    if (n >= N_NODES) return;
    int b = batch[n];
    int st = starts[b], cn = counts[b];
    float my = skey[n];
    int rank = 0;
    for (int j = st; j < st + cn; ++j) {
        float kj = skey[j];
        rank += (kj > my) || (kj == my && j < n);
    }
    int k = (int)ceilf(0.8f * (float)nvalid[b]);
    keep_out[n] = (keep_in[n] && rank < k) ? 1 : 0;
}

__global__ void ggc_xscale(float* __restrict__ h, _Float16* __restrict__ h16,
                           const float* __restrict__ score, const int* __restrict__ keep) {
    int idx = blockIdx.x * blockDim.x + threadIdx.x;
    if (idx >= N_NODES * D_DIM) return;
    int n = idx >> 8;
    float v = keep[n] ? h[idx] * score[n] : 0.f;
    h[idx] = v;
    h16[idx] = (_Float16)v;
}

__global__ void ggc_emask_upd(float* __restrict__ emask, const int* __restrict__ src,
                              const int* __restrict__ dst, const int* __restrict__ keep) {
    int e = blockIdx.x * blockDim.x + threadIdx.x;
    if (e >= E_EDGE) return;
    emask[e] *= (keep[src[e]] && keep[dst[e]]) ? 1.f : 0.f;
}

// ---------------- 6) global attention ----------------
__device__ __forceinline__ unsigned ggc_fmap(float f) {
    unsigned u = __float_as_uint(f);
    return (u & 0x80000000u) ? ~u : (u | 0x80000000u);
}
__device__ __forceinline__ float ggc_funmap(unsigned u) {
    unsigned b = (u & 0x80000000u) ? (u & 0x7FFFFFFFu) : ~u;
    return __uint_as_float(b);
}

__global__ void ggc_logit(const float* __restrict__ h, const float* __restrict__ gW,
                          const float* __restrict__ gb, const int* __restrict__ keep,
                          float* __restrict__ logit) {
    int gid = blockIdx.x * blockDim.x + threadIdx.x;
    int n = gid >> 5, lane = gid & 31;
    if (n >= N_NODES) return;
    const float* row = h + (size_t)n * D_DIM;
    float s = 0.f;
    for (int d = lane; d < D_DIM; d += 32) s += row[d] * gW[d];
    for (int off = 16; off > 0; off >>= 1) s += __shfl_down(s, off, 32);
    if (lane == 0) logit[n] = keep[n] ? (s + gb[0]) : -1e9f;
}

__global__ void ggc_segmax(const int* __restrict__ batch, const float* __restrict__ logit,
                           unsigned* __restrict__ mxu) {
    int n = blockIdx.x * blockDim.x + threadIdx.x;
    if (n >= N_NODES) return;
    atomicMax(&mxu[batch[n]], ggc_fmap(logit[n]));
}

__global__ void ggc_esum(const int* __restrict__ batch, const float* __restrict__ logit,
                         const unsigned* __restrict__ mxu, const int* __restrict__ keep,
                         float* __restrict__ ew, float* __restrict__ den) {
    int n = blockIdx.x * blockDim.x + threadIdx.x;
    if (n >= N_NODES) return;
    int b = batch[n];
    float mx = ggc_funmap(mxu[b]);
    float e = keep[n] ? __expf(logit[n] - mx) : 0.f;
    ew[n] = e;
    if (e != 0.f) atomicAdd(&den[b], e);
}

__global__ void ggc_outacc(const int* __restrict__ batch, const float* __restrict__ h,
                           const float* __restrict__ ew, const float* __restrict__ den,
                           float* __restrict__ out) {
    int idx = blockIdx.x * blockDim.x + threadIdx.x;
    if (idx >= N_NODES * D_DIM) return;
    int n = idx >> 8, d = idx & 255;
    int b = batch[n];
    float g = ew[n] / fmaxf(den[b], 1e-12f);
    if (g != 0.f) atomicAdd(&out[(size_t)b * D_DIM + d], g * h[idx]);
}

// ---------------- host orchestration ----------------
static inline int ggc_nb(long long n) { return (int)((n + 255) / 256); }

extern "C" void kernel_launch(void* const* d_in, const int* in_sizes, int n_in,
                              void* d_out, int out_size, void* d_ws, size_t ws_size,
                              hipStream_t stream) {
    (void)in_sizes; (void)n_in; (void)out_size; (void)ws_size;

    const int*   x_tokens = (const int*)  d_in[0];
    const int*   eidx     = (const int*)  d_in[1];   // [2,E]: src then dst
    const int*   batch    = (const int*)  d_in[2];
    const float* embed    = (const float*)d_in[3];
    const float* ggcW     = (const float*)d_in[4];   // [L,4,D,D]
    const float* Wih      = (const float*)d_in[5];   // [L,3D,D]
    const float* Whh      = (const float*)d_in[6];   // [L,3D,D]
    const float* bih      = (const float*)d_in[7];   // [L,3D]
    const float* bhh      = (const float*)d_in[8];   // [L,3D]
    const float* topk_p   = (const float*)d_in[9];   // [L,D]
    const float* gateW    = (const float*)d_in[10];  // [D,1]
    const float* gateB    = (const float*)d_in[11];  // [1]
    float* out = (float*)d_out;

    const int* src = eidx;
    const int* dst = eidx + E_EDGE;

    const size_t ND = (size_t)N_NODES * D_DIM;
    float* h     = (float*)d_ws;
    float* hw    = h   + ND;
    float* agg   = hw  + ND;
    float* gi    = agg + ND;          // N x 3D
    float* gh    = gi  + 3 * ND;      // N x 3D
    float* emask = gh  + 3 * ND;      // E
    float* score = emask + E_EDGE;    // N
    float* skey  = score + N_NODES;   // N
    float* logit = skey  + N_NODES;   // N
    float* ew    = logit + N_NODES;   // N
    float* pn    = ew    + N_NODES;   // 1
    float* den   = pn    + 1;         // B
    int* keepA   = (int*)(den + B_GR);          // N
    int* keepB   = keepA + N_NODES;             // N
    int* counts  = keepB + N_NODES;             // B
    int* starts  = counts + B_GR;               // B
    int* nvalid  = starts + B_GR;               // B
    unsigned* mxu = (unsigned*)(nvalid + B_GR); // B
    // f16 region (32B aligned)
    size_t foff = (size_t)((char*)(mxu + B_GR) - (char*)d_ws);
    foff = (foff + 31) & ~(size_t)31;
    _Float16* h16    = (_Float16*)((char*)d_ws + foff);                 // N x D
    _Float16* agg16  = h16   + ND;                                      // N x D
    _Float16* wt16   = agg16 + ND;                                      // 12 x D x D ([Dout,K])
    _Float16* wih16  = wt16  + (size_t)N_LAY * N_GRU * D_DIM * D_DIM;   // L x 3D x D
    _Float16* whh16  = wih16 + (size_t)N_LAY * D3_DIM * D_DIM;          // L x 3D x D

    // ---- init + weight prep (f16, B stored [Dout,K]) ----
    ggc_fill_f32<<<ggc_nb(B_GR * D_DIM), 256, 0, stream>>>(out, 0.f, B_GR * D_DIM);
    ggc_fill_f32<<<ggc_nb(E_EDGE), 256, 0, stream>>>(emask, 1.f, E_EDGE);
    ggc_fill_i32<<<ggc_nb(N_NODES), 256, 0, stream>>>(keepA, 1, N_NODES);
    ggc_prep_wt<<<ggc_nb((long long)N_LAY * N_GRU * D_DIM * D_DIM), 256, 0, stream>>>(
        ggcW, wt16, N_LAY * N_GRU);
    ggc_cvt16<<<ggc_nb((long long)N_LAY * D3_DIM * D_DIM), 256, 0, stream>>>(
        Wih, wih16, N_LAY * D3_DIM * D_DIM);
    ggc_cvt16<<<ggc_nb((long long)N_LAY * D3_DIM * D_DIM), 256, 0, stream>>>(
        Whh, whh16, N_LAY * D3_DIM * D_DIM);
    ggc_embed_mean<<<ggc_nb((long long)N_NODES * D_DIM), 256, 0, stream>>>(
        x_tokens, embed, h, h16);

    int* keep_cur = keepA;
    int* keep_nxt = keepB;

    const long long waves_D  = (long long)(N_NODES / 32) * (D_DIM / 64);   // 2500
    const long long waves_3D = (long long)(N_NODES / 32) * (D3_DIM / 64);  // 7500

    for (int l = 0; l < N_LAY; ++l) {
        const _Float16* wih16_l = wih16 + (size_t)l * D3_DIM * D_DIM;
        const _Float16* whh16_l = whh16 + (size_t)l * D3_DIM * D_DIM;
        const float* bih_l = bih + (size_t)l * D3_DIM;
        const float* bhh_l = bhh + (size_t)l * D3_DIM;

        // ---- GatedGraphConv: 4 propagation steps ----
        for (int s = 0; s < N_GRU; ++s) {
            const _Float16* wt16_ls = wt16 + (size_t)(l * N_GRU + s) * D_DIM * D_DIM;
            ggc_gemm_wmma<<<ggc_nb(waves_D * 32), 256, 0, stream>>>(
                h16, wt16_ls, nullptr, hw, N_NODES, D_DIM, D_DIM);
            ggc_fill_f32<<<ggc_nb((long long)ND), 256, 0, stream>>>(agg, 0.f, (int)ND);
            ggc_scatter_add<<<ggc_nb((long long)E_EDGE * (D_DIM / 4)), 256, 0, stream>>>(
                hw, src, dst, emask, agg);
            ggc_cvt16<<<ggc_nb((long long)ND), 256, 0, stream>>>(agg, agg16, (int)ND);
            ggc_gemm_wmma<<<ggc_nb(waves_3D * 32), 256, 0, stream>>>(
                agg16, wih16_l, bih_l, gi, N_NODES, D_DIM, D3_DIM);
            ggc_gemm_wmma<<<ggc_nb(waves_3D * 32), 256, 0, stream>>>(
                h16, whh16_l, bhh_l, gh, N_NODES, D_DIM, D3_DIM);
            ggc_gru_elem<<<ggc_nb((long long)ND), 256, 0, stream>>>(
                gi, gh, h, h16, (s == N_GRU - 1) ? 1 : 0);   // ReLU folded into last step
        }

        // ---- TopK pooling ----
        ggc_pnorm<<<1, 256, 0, stream>>>(topk_p + (size_t)l * D_DIM, pn);
        ggc_score<<<ggc_nb((long long)N_NODES * 32), 256, 0, stream>>>(
            h, topk_p + (size_t)l * D_DIM, pn, keep_cur, score, skey);
        ggc_fill_i32<<<1, 256, 0, stream>>>(counts, 0, B_GR);
        ggc_fill_i32<<<1, 256, 0, stream>>>(nvalid, 0, B_GR);
        ggc_count<<<ggc_nb(N_NODES), 256, 0, stream>>>(batch, keep_cur, counts, nvalid);
        ggc_scan<<<1, 1, 0, stream>>>(counts, starts);
        ggc_rank<<<ggc_nb(N_NODES), 256, 0, stream>>>(
            batch, skey, counts, starts, nvalid, keep_cur, keep_nxt);
        ggc_xscale<<<ggc_nb((long long)ND), 256, 0, stream>>>(h, h16, score, keep_nxt);
        ggc_emask_upd<<<ggc_nb(E_EDGE), 256, 0, stream>>>(emask, src, dst, keep_nxt);

        // ---- Global attention, accumulate into out ----
        ggc_logit<<<ggc_nb((long long)N_NODES * 32), 256, 0, stream>>>(
            h, gateW, gateB, keep_nxt, logit);
        ggc_fill_u32<<<1, 256, 0, stream>>>(mxu, 0u, B_GR);
        ggc_fill_f32<<<1, 256, 0, stream>>>(den, 0.f, B_GR);
        ggc_segmax<<<ggc_nb(N_NODES), 256, 0, stream>>>(batch, logit, mxu);
        ggc_esum<<<ggc_nb(N_NODES), 256, 0, stream>>>(batch, logit, mxu, keep_nxt, ew, den);
        ggc_outacc<<<ggc_nb((long long)ND), 256, 0, stream>>>(batch, h, ew, den, out);

        int* tmp = keep_cur; keep_cur = keep_nxt; keep_nxt = tmp;
    }
}
